// DotProductAttention_16312285790459
// MI455X (gfx1250) — compile-verified, matrix operand order checked by reading, not run on hardware
//
#include <hip/hip_runtime.h>
#include <hip/hip_bf16.h>
#include <math.h>

// Flash-attention (B=8, Nq=Nk=2048, D=64, fp32 in/out) with per-batch key
// masking. f16 WMMA + f32 accumulation, online softmax, key loop truncated at
// valid_len. KT=128 keys/iteration; softmax reductions via DPP (no ds_bpermute);
// V transpose staged as packed b64 stores.

typedef __attribute__((ext_vector_type(16))) _Float16 v16h;
typedef __attribute__((ext_vector_type(8)))  _Float16 v8h;
typedef __attribute__((ext_vector_type(4)))  _Float16 v4h;
typedef __attribute__((ext_vector_type(8)))  float    v8f;

#define NEGVAL (-1000000.0f)

static constexpr int NQ   = 2048;
static constexpr int NK   = 2048;
static constexpr int DK   = 64;
static constexpr int QT   = 64;    // queries per block (4 waves x 16 rows)
static constexpr int KT   = 128;   // keys per main-loop iteration
static constexpr int KSTR = 72;    // stride (halves) for dim-minor tiles (Q,K): 144B, 16B-aligned
static constexpr int VSTR = 136;   // stride (halves) for key-minor tiles (Vt,P): 272B, 16B-aligned

// A- and B-fragments for 16-bit WMMA share the same per-lane pattern:
// lanes 0-15  : halves {kbase+0..7,  kbase+16..23}
// lanes 16-31 : halves {kbase+8..15, kbase+24..31}
// -> two ds_load_b128 at (row*STRIDE + kbase + hsel) and +16 halves.
template <int STRIDE>
__device__ __forceinline__ v16h load_frag(const _Float16* base, int row, int hsel, int kbase) {
    const v8h lo = *(const v8h*)(base + row * STRIDE + kbase + hsel);
    const v8h hi = *(const v8h*)(base + row * STRIDE + kbase + hsel + 16);
    v16h f;
#pragma unroll
    for (int i = 0; i < 8; ++i) { f[i] = lo[i]; f[i + 8] = hi[i]; }
    return f;
}

__device__ __forceinline__ v8f wmma_f16(v16h a, v16h b, v8f c) {
    return __builtin_amdgcn_wmma_f32_16x16x32_f16(false, a, false, b, (short)0, c, false, false);
}

// ---- DPP lane-xor shuffles within a 16-lane half (no DS traffic) ----
__device__ __forceinline__ float dpp_xor1(float x) {   // quad_perm [1,0,3,2]
    return __builtin_bit_cast(float,
        __builtin_amdgcn_update_dpp(0, __builtin_bit_cast(int, x), 0xB1, 0xF, 0xF, true));
}
__device__ __forceinline__ float dpp_xor2(float x) {   // quad_perm [2,3,0,1]
    return __builtin_bit_cast(float,
        __builtin_amdgcn_update_dpp(0, __builtin_bit_cast(int, x), 0x4E, 0xF, 0xF, true));
}
__device__ __forceinline__ float dpp_xor4(float x) {   // dpp8 sel: lane i <- i^4
    constexpr int SEL = 4 | (5 << 3) | (6 << 6) | (7 << 9) |
                        (0 << 12) | (1 << 15) | (2 << 18) | (3 << 21);
    return __builtin_bit_cast(float,
        __builtin_amdgcn_mov_dpp8(__builtin_bit_cast(int, x), SEL));
}
__device__ __forceinline__ float dpp_xor8(float x) {   // ROW_ROR:8 == xor8 within row of 16
    return __builtin_bit_cast(float,
        __builtin_amdgcn_update_dpp(0, __builtin_bit_cast(int, x), 0x128, 0xF, 0xF, true));
}
__device__ __forceinline__ float red16_max(float x) {
    x = fmaxf(x, dpp_xor1(x));
    x = fmaxf(x, dpp_xor2(x));
    x = fmaxf(x, dpp_xor4(x));
    x = fmaxf(x, dpp_xor8(x));
    return x;
}
__device__ __forceinline__ float red16_sum(float x) {
    x += dpp_xor1(x);
    x += dpp_xor2(x);
    x += dpp_xor4(x);
    x += dpp_xor8(x);
    return x;
}

__global__ __launch_bounds__(128) void flash_attn_f16wmma(
    const float* __restrict__ Q, const float* __restrict__ K,
    const float* __restrict__ V, const int* __restrict__ VL,
    float* __restrict__ Out)
{
    __shared__ _Float16 qbuf[QT * KSTR];       // Q tile (query x dim), pre-scaled
    __shared__ _Float16 kbuf[KT * KSTR];       // K tile (key x dim)
    __shared__ _Float16 vtbuf[DK * VSTR];      // V tile transposed (dim x key)
    __shared__ _Float16 pbuf[4 * 16 * VSTR];   // per-wave P scratch (C->A relayout)

    const int tid  = threadIdx.x;
    const int wid  = tid >> 5;
    const int lane = tid & 31;
    const int ln16 = lane & 15;
    const int hsel = (lane >> 4) << 3;         // 0 or 8

    const int b  = blockIdx.x >> 5;            // 32 query-blocks per batch
    const int qb = (blockIdx.x & 31) * QT;

    int vlen = VL[b];
    if (vlen < 1)  vlen = 1;
    if (vlen > NK) vlen = NK;
    const int nkb = (vlen + KT - 1) / KT;      // fully-masked tail blocks skipped

    // ---- stage Q tile, f32 -> f16, fold in 1/sqrt(64) ----
    for (int i = tid; i < QT * DK / 4; i += 128) {
        const int q  = i >> 4;
        const int d4 = (i & 15) << 2;
        const float4 f = *(const float4*)(Q + (((size_t)b * NQ + qb + q) * DK + d4));
        v4h h;
        h[0] = (_Float16)(f.x * 0.125f); h[1] = (_Float16)(f.y * 0.125f);
        h[2] = (_Float16)(f.z * 0.125f); h[3] = (_Float16)(f.w * 0.125f);
        *(v4h*)(qbuf + q * KSTR + d4) = h;
    }
    __syncthreads();

    // per-wave Q A-fragments, kept in registers for the whole kernel
    const int qrow = wid * 16 + ln16;
    const v16h qa0 = load_frag<KSTR>(qbuf, qrow, hsel, 0);
    const v16h qa1 = load_frag<KSTR>(qbuf, qrow, hsel, 32);

    v8f o0 = {}, o1 = {}, o2 = {}, o3 = {};    // O accumulator: 16 x 64 f32
    float m[8], l[8];
#pragma unroll
    for (int i = 0; i < 8; ++i) { m[i] = -3.0e38f; l[i] = 0.0f; }

    _Float16* pw = pbuf + wid * 16 * VSTR;

    for (int kb = 0; kb < nkb; ++kb) {
        __syncthreads();                       // previous block's kbuf/vtbuf reads done

        const size_t kgbase = ((size_t)b * NK + (size_t)kb * KT) * DK;
        // ---- stage K (key x dim), f32 -> f16, b64 stores ----
        for (int i = tid; i < KT * DK / 4; i += 128) {
            const int r  = i >> 4;
            const int d4 = (i & 15) << 2;
            const float4 fk = *(const float4*)(K + kgbase + r * DK + d4);
            v4h hk;
            hk[0] = (_Float16)fk.x; hk[1] = (_Float16)fk.y;
            hk[2] = (_Float16)fk.z; hk[3] = (_Float16)fk.w;
            *(v4h*)(kbuf + r * KSTR + d4) = hk;
        }
        // ---- stage V transposed via 4x4 register micro-tiles, b64 stores ----
        for (int u = tid; u < (KT / 4) * (DK / 4); u += 128) {
            const int k0 = (u >> 4) << 2;      // key group of 4
            const int d0 = (u & 15) << 2;      // dim group of 4
            const float4 r0 = *(const float4*)(V + kgbase + (size_t)(k0 + 0) * DK + d0);
            const float4 r1 = *(const float4*)(V + kgbase + (size_t)(k0 + 1) * DK + d0);
            const float4 r2 = *(const float4*)(V + kgbase + (size_t)(k0 + 2) * DK + d0);
            const float4 r3 = *(const float4*)(V + kgbase + (size_t)(k0 + 3) * DK + d0);
            v4h c0 = { (_Float16)r0.x, (_Float16)r1.x, (_Float16)r2.x, (_Float16)r3.x };
            v4h c1 = { (_Float16)r0.y, (_Float16)r1.y, (_Float16)r2.y, (_Float16)r3.y };
            v4h c2 = { (_Float16)r0.z, (_Float16)r1.z, (_Float16)r2.z, (_Float16)r3.z };
            v4h c3 = { (_Float16)r0.w, (_Float16)r1.w, (_Float16)r2.w, (_Float16)r3.w };
            *(v4h*)(vtbuf + (d0 + 0) * VSTR + k0) = c0;
            *(v4h*)(vtbuf + (d0 + 1) * VSTR + k0) = c1;
            *(v4h*)(vtbuf + (d0 + 2) * VSTR + k0) = c2;
            *(v4h*)(vtbuf + (d0 + 3) * VSTR + k0) = c3;
        }
        __syncthreads();

        // ---- scores S = (Q/8) K^T : 16 x 128, 16 WMMAs ----
        v8f s[8];
#pragma unroll
        for (int nt = 0; nt < 8; ++nt) {
            v8f acc = {};
            const v16h kf0 = load_frag<KSTR>(kbuf, nt * 16 + ln16, hsel, 0);
            acc = wmma_f16(qa0, kf0, acc);
            const v16h kf1 = load_frag<KSTR>(kbuf, nt * 16 + ln16, hsel, 32);
            acc = wmma_f16(qa1, kf1, acc);
            s[nt] = acc;
        }

        // ---- mask tail keys in the final block (uniform branch; EXEC stays full) ----
        if (kb == nkb - 1) {
#pragma unroll
            for (int nt = 0; nt < 8; ++nt) {
                const bool mk = (kb * KT + nt * 16 + ln16) >= vlen;
#pragma unroll
                for (int i = 0; i < 8; ++i) s[nt][i] = mk ? NEGVAL : s[nt][i];
            }
        }

        // ---- online softmax; row stats live in 16-lane halves (C-layout) ----
#pragma unroll
        for (int i = 0; i < 8; ++i) {
            float rm = s[0][i];
#pragma unroll
            for (int nt = 1; nt < 8; ++nt) rm = fmaxf(rm, s[nt][i]);
            rm = red16_max(rm);
            const float mnew  = fmaxf(m[i], rm);
            const float alpha = __expf(m[i] - mnew);
            m[i] = mnew;
            float rs = 0.0f;
#pragma unroll
            for (int nt = 0; nt < 8; ++nt) {
                s[nt][i] = __expf(s[nt][i] - mnew);
                rs += s[nt][i];
            }
            rs = red16_sum(rs);
            l[i] = l[i] * alpha + rs;
            o0[i] *= alpha; o1[i] *= alpha; o2[i] *= alpha; o3[i] *= alpha;
        }

        // ---- P: C-layout -> A-layout via per-wave LDS scratch (f16) ----
#pragma unroll
        for (int i = 0; i < 8; ++i) {
            const int row = i + ((lane >> 4) << 3);
#pragma unroll
            for (int nt = 0; nt < 8; ++nt)
                pw[row * VSTR + nt * 16 + ln16] = (_Float16)s[nt][i];
        }
        // DS ops from one wave are in-order; compiler inserts s_wait_dscnt for reuse.

        // ---- O += P V : 16 WMMAs (B-frags read contiguous keys from Vt) ----
#pragma unroll
        for (int h = 0; h < 4; ++h) {
            const v16h pf = load_frag<VSTR>(pw, ln16, hsel, h * 32);
            v16h vf;
            vf = load_frag<VSTR>(vtbuf,  0 + ln16, hsel, h * 32); o0 = wmma_f16(pf, vf, o0);
            vf = load_frag<VSTR>(vtbuf, 16 + ln16, hsel, h * 32); o1 = wmma_f16(pf, vf, o1);
            vf = load_frag<VSTR>(vtbuf, 32 + ln16, hsel, h * 32); o2 = wmma_f16(pf, vf, o2);
            vf = load_frag<VSTR>(vtbuf, 48 + ln16, hsel, h * 32); o3 = wmma_f16(pf, vf, o3);
        }
    }

    // ---- epilogue: O /= l, store fp32 ----
#pragma unroll
    for (int i = 0; i < 8; ++i) {
        const float inv = 1.0f / l[i];
        const int row = i + ((lane >> 4) << 3);
        float* op = Out + (((size_t)b * NQ + qb + wid * 16 + row) * DK + ln16);
        op[0]  = o0[i] * inv;
        op[16] = o1[i] * inv;
        op[32] = o2[i] * inv;
        op[48] = o3[i] * inv;
    }
}

extern "C" void kernel_launch(void* const* d_in, const int* in_sizes, int n_in,
                              void* d_out, int out_size, void* d_ws, size_t ws_size,
                              hipStream_t stream) {
    (void)in_sizes; (void)n_in; (void)out_size; (void)d_ws; (void)ws_size;
    const float* Q  = (const float*)d_in[0];
    const float* K  = (const float*)d_in[1];
    const float* V  = (const float*)d_in[2];
    const int*   VL = (const int*)d_in[3];
    float* Out = (float*)d_out;
    const int B = 8;
    dim3 grid(B * (NQ / QT));   // 256 blocks
    dim3 block(128);            // 4 waves
    flash_attn_f16wmma<<<grid, block, 0, stream>>>(Q, K, V, VL, Out);
}
